// HybridReadout_47072841564533
// MI455X (gfx1250) — matrix operand order repaired and checked by reading, not compiled, not run on hardware
//
#include <hip/hip_runtime.h>
#include <hip/hip_bf16.h>

typedef __attribute__((ext_vector_type(16))) _Float16 v16h;
typedef __attribute__((ext_vector_type(8)))  _Float16 v8h;
typedef __attribute__((ext_vector_type(4)))  _Float16 v4h;
typedef __attribute__((ext_vector_type(8)))  float    v8f;

#define N_ROWS 4096
#define H_DIM  4096
#define O_DIM  32000
#define K_TOT  8192   // 2*H: [0.6*spikes | 0.4*membranes]
#define BM 128
#define BN 256
#define BK 32
#define LDT 40        // padded LDS row stride (halfs)

#if __has_builtin(__builtin_amdgcn_global_load_async_to_lds_b128)
#define HAVE_ASYNC_LDS 1
#else
#define HAVE_ASYNC_LDS 0
#endif

__device__ __forceinline__ void wait_async_zero()
{
#if HAVE_ASYNC_LDS
#if __has_builtin(__builtin_amdgcn_s_wait_asynccnt)
    __builtin_amdgcn_s_wait_asynccnt(0);
#else
    asm volatile("s_wait_asynccnt 0x0" ::: "memory");
#endif
#endif
}

// 16B global -> LDS copy: async (ASYNCcnt, no VGPR data) when available.
// Builtin signature (from hipcc diagnostic): (int4_vs __device__*, int4_vs __shared__*, imm, imm)
#if HAVE_ASYNC_LDS
typedef int v4i_vs __attribute__((vector_size(16)));
__device__ __forceinline__ void cp_b128(const _Float16* g, _Float16* l)
{
    __builtin_amdgcn_global_load_async_to_lds_b128(
        (__attribute__((address_space(1))) v4i_vs*)g,
        (__attribute__((address_space(3))) v4i_vs*)l,
        0, 0);
}
#else
__device__ __forceinline__ void cp_b128(const _Float16* g, _Float16* l)
{
    *(uint4*)l = *(const uint4*)g;
}
#endif

// ---------------------------------------------------------------------------
// Kernel 1: sum(|w|) over a weight matrix (float4 grid-stride + block reduce)
// ---------------------------------------------------------------------------
__global__ void absum_kernel(const float* __restrict__ w,
                             unsigned long long n4,
                             float* __restrict__ out)
{
    __shared__ float red[256];
    float s = 0.0f;
    unsigned long long idx    = (unsigned long long)blockIdx.x * blockDim.x + threadIdx.x;
    unsigned long long stride = (unsigned long long)gridDim.x * blockDim.x;
    const float4* w4 = (const float4*)w;
    for (unsigned long long c = idx; c < n4; c += stride) {
        float4 v = w4[c];
        s += fabsf(v.x) + fabsf(v.y) + fabsf(v.z) + fabsf(v.w);
    }
    red[threadIdx.x] = s;
    __syncthreads();
    for (int off = 128; off > 0; off >>= 1) {
        if ((int)threadIdx.x < off) red[threadIdx.x] += red[threadIdx.x + off];
        __syncthreads();
    }
    if (threadIdx.x == 0) atomicAdd(out, red[0]);
}

// ---------------------------------------------------------------------------
// Kernel 2: pack activations into f16 [N_ROWS, K_TOT] = [0.6*S | 0.4*M]
// ---------------------------------------------------------------------------
__global__ void pack_act_kernel(const float* __restrict__ spikes,
                                const float* __restrict__ membranes,
                                _Float16* __restrict__ Ap)
{
    const unsigned long long total = (unsigned long long)N_ROWS * K_TOT / 4;
    unsigned long long idx    = (unsigned long long)blockIdx.x * blockDim.x + threadIdx.x;
    unsigned long long stride = (unsigned long long)gridDim.x * blockDim.x;
    for (unsigned long long c = idx; c < total; c += stride) {
        int n  = (int)(c >> 11);        // K_TOT/4 = 2048 chunks per row
        int k  = ((int)(c & 2047)) * 4;
        float4 v; float s;
        if (k < H_DIM) { v = *(const float4*)(spikes    + (size_t)n * H_DIM + k);           s = 0.6f; }
        else           { v = *(const float4*)(membranes + (size_t)n * H_DIM + (k - H_DIM)); s = 0.4f; }
        v4h h;
        h.x = (_Float16)(v.x * s);
        h.y = (_Float16)(v.y * s);
        h.z = (_Float16)(v.z * s);
        h.w = (_Float16)(v.w * s);
        *(v4h*)(Ap + (size_t)n * K_TOT + k) = h;
    }
}

// ternary quantize: sign(w) if |w| > t else 0 (exact in f16)
__device__ __forceinline__ _Float16 tq(float w, float t)
{
    return (_Float16)((fabsf(w) > t) ? copysignf(1.0f, w) : 0.0f);
}

// ---------------------------------------------------------------------------
// Kernel 3: WMMA GEMM.  C[4096,32000] = Ap[4096,8192] @ Wcat[32000,8192]^T
// 256 thr = 8 wave32; block tile 128x256; wave tile 64x64 (4x4 frags);
// double-buffered LDS, async A-tile loads, on-the-fly ternary quantize of W.
// ---------------------------------------------------------------------------
__global__ __launch_bounds__(256)
void ternary_wmma_gemm(const _Float16* __restrict__ Ap,
                       const float* __restrict__ w_spike,
                       const float* __restrict__ w_membrane,
                       const float* __restrict__ sums,
                       float* __restrict__ out)
{
    __shared__ _Float16 As[2][BM * LDT];   // 2 x 10 KB
    __shared__ _Float16 Bs[2][BN * LDT];   // 2 x 20 KB

    const int tid   = threadIdx.x;
    const int lane  = tid & 31;
    const int wave  = tid >> 5;
    const int waveM = wave & 1;    // 2 x 64 rows
    const int waveN = wave >> 1;   // 4 x 64 cols
    const int lhi   = lane >> 4;
    const int llo   = lane & 15;

    const int rowBlock = blockIdx.x * BM;
    const int colBlock = blockIdx.y * BN;

    const float inv = 1.0f / ((float)O_DIM * (float)H_DIM);
    const float tS  = 0.7f * sums[0] * inv;
    const float tM  = 0.7f * sums[1] * inv;

    v8f acc[4][4];
    #pragma unroll
    for (int mf = 0; mf < 4; ++mf)
        #pragma unroll
        for (int nf = 0; nf < 4; ++nf)
            acc[mf][nf] = (v8f){0.f, 0.f, 0.f, 0.f, 0.f, 0.f, 0.f, 0.f};

    float4 wreg[8];

    // A tile: 128x32 halfs = 512 x 16B chunks, async to LDS
    auto stageA = [&](int b, int kbase) {
        #pragma unroll
        for (int i = 0; i < 2; ++i) {
            int c  = tid + 256 * i;
            int r  = c >> 2;
            int kc = (c & 3) * 8;
            cp_b128(Ap + (size_t)(rowBlock + r) * K_TOT + kbase + kc,
                    &As[b][r * LDT + kc]);
        }
    };
    // W tile: 256x32 fp32 loads into registers (K-tile never straddles H)
    auto loadW = [&](int kbase) {
        const float* wsrc; int koff;
        if (kbase < H_DIM) { wsrc = w_spike;    koff = kbase;         }
        else               { wsrc = w_membrane; koff = kbase - H_DIM; }
        #pragma unroll
        for (int i = 0; i < 8; ++i) {
            int c  = tid + 256 * i;
            int r  = c >> 3;
            int kc = (c & 7) * 4;
            wreg[i] = *(const float4*)(wsrc + (size_t)(colBlock + r) * H_DIM + koff + kc);
        }
    };
    // quantize staged W registers into LDS buffer b
    auto storeW = [&](int b, int kbase) {
        const float th = (kbase < H_DIM) ? tS : tM;
        #pragma unroll
        for (int i = 0; i < 8; ++i) {
            int c  = tid + 256 * i;
            int r  = c >> 3;
            int kc = (c & 7) * 4;
            v4h q;
            q.x = tq(wreg[i].x, th);
            q.y = tq(wreg[i].y, th);
            q.z = tq(wreg[i].z, th);
            q.w = tq(wreg[i].w, th);
            *(v4h*)(&Bs[b][r * LDT + kc]) = q;
        }
    };
    // 16 WMMAs from buffer b (CDNA5 wave32 fragment layouts)
    auto compute = [&](int b) {
        v16h aF[4];
        #pragma unroll
        for (int mf = 0; mf < 4; ++mf) {
            int r = waveM * 64 + mf * 16 + llo;    // A: row = lane&15
            int o = lhi * 8;                       // K halves at o, o+16
            v8h lo8 = *(const v8h*)(&As[b][r * LDT + o]);
            v8h hi8 = *(const v8h*)(&As[b][r * LDT + o + 16]);
            aF[mf] = __builtin_shufflevector(lo8, hi8,
                      0,1,2,3,4,5,6,7,8,9,10,11,12,13,14,15);
        }
        #pragma unroll
        for (int nf = 0; nf < 4; ++nf) {
            int n = waveN * 64 + nf * 16 + llo;    // B: col = lane&15
            int o = lhi * 8;                       // K halves at o, o+16
            v8h lo8 = *(const v8h*)(&Bs[b][n * LDT + o]);
            v8h hi8 = *(const v8h*)(&Bs[b][n * LDT + o + 16]);
            v16h bF = __builtin_shufflevector(lo8, hi8,
                       0,1,2,3,4,5,6,7,8,9,10,11,12,13,14,15);
            #pragma unroll
            for (int mf = 0; mf < 4; ++mf)
                acc[mf][nf] = __builtin_amdgcn_wmma_f32_16x16x32_f16(
                    false, aF[mf], false, bF,
                    (short)0, acc[mf][nf], false, false);
        }
    };

    // ---- prologue: fill buffer 0 ----
    stageA(0, 0);
    loadW(0);
    storeW(0, 0);
    wait_async_zero();
    __syncthreads();

    // ---- main loop: single barrier per K-step, next tile in flight ----
    int buf = 0;
    for (int kb = 0; kb < K_TOT; kb += BK) {
        const int  nxt     = buf ^ 1;
        const bool hasNext = (kb + BK) < K_TOT;

        if (hasNext) {
            stageA(nxt, kb + BK);     // async: LDS direct, no VGPRs
            loadW(kb + BK);           // global latency hides under WMMAs
        }

        compute(buf);

        if (hasNext) storeW(nxt, kb + BK);

        wait_async_zero();            // my async writes to nxt complete
        __syncthreads();              // all waves' writes visible
        buf = nxt;
    }

    // ---- epilogue: C layout VGPR r -> M = r + 8*(lane>>4), N = lane&15 ----
    #pragma unroll
    for (int mf = 0; mf < 4; ++mf) {
        #pragma unroll
        for (int nf = 0; nf < 4; ++nf) {
            int col   = colBlock + waveN * 64 + nf * 16 + llo;
            int rbase = rowBlock + waveM * 64 + mf * 16 + lhi * 8;
            #pragma unroll
            for (int r = 0; r < 8; ++r)
                out[(size_t)(rbase + r) * O_DIM + col] = acc[mf][nf][r];
        }
    }
}

// ---------------------------------------------------------------------------
extern "C" void kernel_launch(void* const* d_in, const int* in_sizes, int n_in,
                              void* d_out, int out_size, void* d_ws, size_t ws_size,
                              hipStream_t stream)
{
    (void)in_sizes; (void)n_in; (void)out_size; (void)ws_size;

    const float* spikes     = (const float*)d_in[0];
    const float* membranes  = (const float*)d_in[1];
    const float* w_spike    = (const float*)d_in[2];
    const float* w_membrane = (const float*)d_in[3];
    float* out = (float*)d_out;

    // workspace: [0..255] abs-sum accumulators; then 64 MB packed activations
    float*    sums = (float*)d_ws;
    _Float16* Ap   = (_Float16*)((char*)d_ws + 256);

    (void)hipMemsetAsync(d_ws, 0, 256, stream);

    const unsigned long long nW4 = (unsigned long long)O_DIM * H_DIM / 4;
    absum_kernel<<<4096, 256, 0, stream>>>(w_spike,    nW4, sums + 0);
    absum_kernel<<<4096, 256, 0, stream>>>(w_membrane, nW4, sums + 1);

    pack_act_kernel<<<8192, 256, 0, stream>>>(spikes, membranes, Ap);

    dim3 grid(N_ROWS / BM, O_DIM / BN);   // 32 x 125
    ternary_wmma_gemm<<<grid, 256, 0, stream>>>(Ap, w_spike, w_membrane, sums, out);
}
